// MPN_3092376453604
// MI455X (gfx1250) — matrix operand order repaired and checked by reading, not compiled, not run on hardware
//
#include <hip/hip_runtime.h>
#include <cstddef>

// ---------------- types for WMMA ----------------
typedef __attribute__((ext_vector_type(16))) __bf16 v16bf;
typedef __attribute__((ext_vector_type(8)))  float  v8f;

union ABPack {
    uint4 q[2];   // 32 bytes
    v16bf v;      // 16 bf16
};

// ---------------- constants ----------------
#define HIDDEN 300
#define N_MOLS 4096
#define APM    32
#define N_ATOMS (N_MOLS * APM)      // 131072
#define N_BONDS (2 * N_ATOMS)       // 262144
#define ROW_F4 (HIDDEN / 4)         // 75
#define NT     20                   // N subtiles of 16 covering 320 (pad of 300)
#define NTILE  160                  // per-block N tile (10 subtiles), grid.y = 2
#define AS_STRIDE 40                // padded LDS stride (halves), keeps 16B align

__device__ __forceinline__ unsigned short f2bf(float f) {
    unsigned int u = __float_as_uint(f);
    unsigned int r = u + 0x7FFFu + ((u >> 16) & 1u);   // round-to-nearest-even
    return (unsigned short)(r >> 16);
}

// =====================================================================
// Weight pre-pack: W[KxN] f32 -> bf16 fragments in this file's contiguous-K
// V_WMMA_16X16X32_BF16 B-operand lane layout.
// One thread per (kt, nt, lane); each writes 16 halves (32B).
//   lane<16 : col = nt*16+lane,     K = kt*32 + e       (e = 0..15)
//   lane>=16: col = nt*16+lane-16,  K = kt*32 + 16 + e
// =====================================================================
__global__ __launch_bounds__(256)
void pack_weights(const float* __restrict__ W, unsigned short* __restrict__ P,
                  int K, int N, int KT)
{
    int t = blockIdx.x * blockDim.x + threadIdx.x;
    if (t >= KT * NT * 32) return;
    int lane = t & 31;
    int nt   = (t >> 5) % NT;
    int kt   = t / (NT * 32);
    int col  = nt * 16 + (lane & 15);
    int kb   = kt * 32 + ((lane >> 4) ? 16 : 0);

    unsigned short vals[16];
    #pragma unroll
    for (int e = 0; e < 16; ++e) {
        int k = kb + e;
        float f = (k < K && col < N) ? W[(size_t)k * N + col] : 0.f;
        vals[e] = f2bf(f);
    }
    uint4 lo, hi;
    lo.x = (unsigned)vals[1]  << 16 | vals[0];
    lo.y = (unsigned)vals[3]  << 16 | vals[2];
    lo.z = (unsigned)vals[5]  << 16 | vals[4];
    lo.w = (unsigned)vals[7]  << 16 | vals[6];
    hi.x = (unsigned)vals[9]  << 16 | vals[8];
    hi.y = (unsigned)vals[11] << 16 | vals[10];
    hi.z = (unsigned)vals[13] << 16 | vals[12];
    hi.w = (unsigned)vals[15] << 16 | vals[14];
    uint4* dst = (uint4*)(P + (size_t)t * 16);
    dst[0] = lo;
    dst[1] = hi;
}

// =====================================================================
// bf16-WMMA GEMM:  P = A[MxK] @ W[KxN]  (+Cin) (+bias)
//   A is f32 (ABF16=false) or bf16 (ABF16=true), row-major, lda=K
//   W comes pre-packed in B-fragment layout (packB)
//   OutRaw (f32)  <- P                     (if non-null)
//   OutAct32      <- relu(P)  f32          (if non-null)
//   OutAct16      <- relu(P)  bf16         (if non-null)
// Block: 256 threads (8 waves). Tile: 128 rows x 160 cols. K-step 32.
// Software-pipelined: next A K-tile global loads overlap the WMMA burst,
// B fragments double-rotated so the 10 WMMAs issue back-to-back.
// =====================================================================
template <bool ABF16>
__global__ __launch_bounds__(256)
void gemm_wmma(const void* __restrict__ Av, const unsigned short* __restrict__ packB,
               const float* __restrict__ Cin, const float* __restrict__ bias,
               float* __restrict__ OutRaw, float* __restrict__ OutAct32,
               unsigned short* __restrict__ OutAct16,
               int M, int K, int N, int KT)
{
    const float*          A32 = (const float*)Av;
    const unsigned short* A16 = (const unsigned short*)Av;

    __shared__ unsigned short As[128 * AS_STRIDE];

    const int tid   = threadIdx.x;
    const int lane  = tid & 31;
    const int wave  = tid >> 5;
    const int mBase = blockIdx.x * 128;
    const int nBase = blockIdx.y * NTILE;
    const int ntg0  = blockIdx.y * 10;       // first global N subtile

    const int l15   = lane & 15;
    const int hi    = lane >> 4;             // 0 | 1
    const int arow  = wave * 16 + l15;
    const int khalf = hi ? 16 : 0;           // A frag K halves select

    v8f acc[10] = {};

    // ---- prefetch A K-tile 0 into registers ----
    float4  r32[4];
    ushort4 r16[4];
    #pragma unroll
    for (int i = 0; i < 4; ++i) {
        int f = tid + 256 * i;               // 0..1023
        int r = f >> 3, qd = f & 7;
        int gk = qd * 4;
        if (ABF16) {
            ushort4 v = {0, 0, 0, 0};
            if (gk < K) v = *(const ushort4*)(A16 + (size_t)(mBase + r) * K + gk);
            r16[i] = v;
        } else {
            float4 v = make_float4(0.f, 0.f, 0.f, 0.f);
            if (gk < K) v = *(const float4*)(A32 + (size_t)(mBase + r) * K + gk);
            r32[i] = v;
        }
    }

    for (int kt = 0; kt < KT; ++kt) {
        __syncthreads();                     // previous tile fully consumed
        // ---- commit prefetched A tile to LDS (bf16) ----
        #pragma unroll
        for (int i = 0; i < 4; ++i) {
            int f = tid + 256 * i;
            int r = f >> 3, qd = f & 7;
            ushort4 h;
            if (ABF16) {
                h = r16[i];
            } else {
                h.x = f2bf(r32[i].x); h.y = f2bf(r32[i].y);
                h.z = f2bf(r32[i].z); h.w = f2bf(r32[i].w);
            }
            *(ushort4*)&As[r * AS_STRIDE + qd * 4] = h;
        }
        __syncthreads();

        // ---- prefetch next A K-tile (overlaps WMMA burst below) ----
        if (kt + 1 < KT) {
            int k0n = (kt + 1) * 32;
            #pragma unroll
            for (int i = 0; i < 4; ++i) {
                int f = tid + 256 * i;
                int r = f >> 3, qd = f & 7;
                int gk = k0n + qd * 4;
                if (ABF16) {
                    ushort4 v = {0, 0, 0, 0};
                    if (gk < K) v = *(const ushort4*)(A16 + (size_t)(mBase + r) * K + gk);
                    r16[i] = v;
                } else {
                    float4 v = make_float4(0.f, 0.f, 0.f, 0.f);
                    if (gk < K) v = *(const float4*)(A32 + (size_t)(mBase + r) * K + gk);
                    r32[i] = v;
                }
            }
        }

        // ---- A fragment from LDS ----
        ABPack ap;
        ap.q[0] = *(const uint4*)&As[arow * AS_STRIDE + khalf];
        ap.q[1] = *(const uint4*)&As[arow * AS_STRIDE + khalf + 8];

        // ---- 10 WMMAs; B fragments streamed from packed weights (L2-hot) ----
        const unsigned short* pb =
            packB + (((size_t)kt * NT + ntg0) * 32 + lane) * 16;
        ABPack bc, bn;
        bc.q[0] = *(const uint4*)pb;
        bc.q[1] = *(const uint4*)(pb + 8);
        #pragma unroll
        for (int j = 0; j < 10; ++j) {
            if (j < 9) {
                const unsigned short* pn = pb + (size_t)(j + 1) * (32 * 16);
                bn.q[0] = *(const uint4*)pn;
                bn.q[1] = *(const uint4*)(pn + 8);
            }
            acc[j] = __builtin_amdgcn_wmma_f32_16x16x32_bf16(
                false, ap.v, false, bc.v, (short)0, acc[j], false, false);
            if (j < 9) bc = bn;
        }
    }

    // ---- epilogue: VGPR r -> M = r (+8 for lanes>=16), N = lane&15 ----
    #pragma unroll
    for (int j = 0; j < 10; ++j) {
        int col = nBase + j * 16 + l15;
        if (col < N) {
            float bv = bias ? bias[col] : 0.f;
            #pragma unroll
            for (int r = 0; r < 8; ++r) {
                int row = mBase + wave * 16 + (hi ? r + 8 : r);
                float v = acc[j][r] + bv;
                if (Cin)      v += Cin[(size_t)row * N + col];
                if (OutRaw)   OutRaw[(size_t)row * N + col] = v;
                float va = v > 0.f ? v : 0.f;
                if (OutAct32) OutAct32[(size_t)row * N + col] = va;
                if (OutAct16) OutAct16[(size_t)row * N + col] = f2bf(va);
            }
        }
    }
}

// =====================================================================
// Gather-sum over bf16 rows: out[b,:] = bf16(sum_{k<6} src[graph[b][k],:])
// 300 halves = 75 x uint2 (4 values) per row; f32 accumulation.
// =====================================================================
__global__ __launch_bounds__(256)
void gather_sum6_bf16(const unsigned short* __restrict__ src,
                      const int* __restrict__ graph,
                      unsigned short* __restrict__ out, int rows)
{
    long t = (long)blockIdx.x * blockDim.x + threadIdx.x;
    if (t >= (long)rows * ROW_F4) return;
    int b  = (int)(t / ROW_F4);
    int qd = (int)(t % ROW_F4);
    const int* g = graph + (size_t)b * 6;
    float s0 = 0.f, s1 = 0.f, s2 = 0.f, s3 = 0.f;
    #pragma unroll
    for (int k = 0; k < 6; ++k) {
        const uint2* p = (const uint2*)(src + (size_t)g[k] * HIDDEN);
        uint2 v = p[qd];
        s0 += __uint_as_float(v.x << 16);
        s1 += __uint_as_float(v.x & 0xffff0000u);
        s2 += __uint_as_float(v.y << 16);
        s3 += __uint_as_float(v.y & 0xffff0000u);
    }
    uint2 o;
    o.x = ((unsigned)f2bf(s1) << 16) | f2bf(s0);
    o.y = ((unsigned)f2bf(s3) << 16) | f2bf(s2);
    ((uint2*)out)[t] = o;
}

// =====================================================================
// Per-molecule self-attention: att0 = softmax(q h^T) h   (h,q: [32,300] f32)
// one block per molecule; h staged in LDS (~42.6 KB)
// =====================================================================
__global__ __launch_bounds__(256)
void attention_kernel(const float* __restrict__ atom_h, const float* __restrict__ q,
                      float* __restrict__ att0)
{
    __shared__ float hS[APM][HIDDEN];
    __shared__ float sS[APM][APM + 1];

    const int m = blockIdx.x;
    const float* hb = atom_h + (size_t)m * APM * HIDDEN;
    const float* qb = q      + (size_t)m * APM * HIDDEN;

    for (int i = threadIdx.x; i < APM * HIDDEN; i += blockDim.x)
        hS[i / HIDDEN][i % HIDDEN] = hb[i];
    __syncthreads();

    for (int i = threadIdx.x; i < APM * APM; i += blockDim.x) {
        int a = i >> 5, c = i & 31;
        const float* qr = qb + (size_t)a * HIDDEN;
        float s = 0.f;
        for (int k = 0; k < HIDDEN; ++k) s += qr[k] * hS[c][k];
        sS[a][c] = s;
    }
    __syncthreads();

    if (threadIdx.x < APM) {
        int a = threadIdx.x;
        float mx = sS[a][0];
        for (int c = 1; c < APM; ++c) mx = fmaxf(mx, sS[a][c]);
        float sum = 0.f;
        for (int c = 0; c < APM; ++c) { float e = __expf(sS[a][c] - mx); sS[a][c] = e; sum += e; }
        float inv = 1.f / sum;
        for (int c = 0; c < APM; ++c) sS[a][c] *= inv;
    }
    __syncthreads();

    float* ob = att0 + (size_t)m * APM * HIDDEN;
    for (int i = threadIdx.x; i < APM * HIDDEN; i += blockDim.x) {
        int a = i / HIDDEN, j = i % HIDDEN;
        float s = 0.f;
        #pragma unroll
        for (int c = 0; c < APM; ++c) s += sS[a][c] * hS[c][j];
        ob[i] = s;
    }
}

// =====================================================================
// Pooling: mol_vecs[m] = sum_a (atom_h[m,a,:] + att_hb[m,a,:]) / 32
// =====================================================================
__global__ __launch_bounds__(256)
void pool_kernel(const float* __restrict__ atom_h, const float* __restrict__ att_hb,
                 float* __restrict__ out)
{
    int t = blockIdx.x * blockDim.x + threadIdx.x;   // over 4096*75
    if (t >= N_MOLS * ROW_F4) return;
    int m  = t / ROW_F4;
    int qd = t % ROW_F4;
    const float4* h4 = (const float4*)(atom_h + (size_t)m * APM * HIDDEN);
    const float4* a4 = (const float4*)(att_hb + (size_t)m * APM * HIDDEN);
    float4 s = make_float4(0.f, 0.f, 0.f, 0.f);
    #pragma unroll 4
    for (int a = 0; a < APM; ++a) {
        float4 x = h4[a * ROW_F4 + qd];
        float4 y = a4[a * ROW_F4 + qd];
        s.x += x.x + y.x; s.y += x.y + y.y; s.z += x.z + y.z; s.w += x.w + y.w;
    }
    const float inv = 1.f / APM;
    s.x *= inv; s.y *= inv; s.z *= inv; s.w *= inv;
    ((float4*)out)[t] = s;
}

// =====================================================================
extern "C" void kernel_launch(void* const* d_in, const int* in_sizes, int n_in,
                              void* d_out, int out_size, void* d_ws, size_t ws_size,
                              hipStream_t stream)
{
    const float* fatoms = (const float*)d_in[0];
    const float* fbonds = (const float*)d_in[1];
    const int*   agraph = (const int*)d_in[2];
    const int*   bgraph = (const int*)d_in[3];
    const float* W_i    = (const float*)d_in[4];
    const float* W_h    = (const float*)d_in[5];
    const float* W_o    = (const float*)d_in[6];
    const float* b_o    = (const float*)d_in[7];
    const float* W_a    = (const float*)d_in[8];
    const float* W_b    = (const float*)d_in[9];
    const float* b_b    = (const float*)d_in[10];
    float* out = (float*)d_out;

    const size_t NBH = (size_t)N_BONDS * HIDDEN;   // 78,643,200
    const size_t NAH = (size_t)N_ATOMS * HIDDEN;   // 39,321,600

    // -------- workspace layout --------
    char* wsb = (char*)d_ws;
    size_t off = 0;
    float*          binput   = (float*)(wsb + off);          off += NBH * 4;  // f32
    unsigned short* message  = (unsigned short*)(wsb + off); off += NBH * 2;  // bf16
    unsigned short* nei      = (unsigned short*)(wsb + off); off += NBH * 2;  // bf16
    float*          scratch0 = (float*)(wsb + off);          off += NAH * 4;  // tmpO / att_hb
    float*          scratch1 = (float*)(wsb + off);          off += NAH * 4;  // atom_h
    unsigned short* packs    = (unsigned short*)(wsb + off);

    const size_t FRAG = 32 * 16;                   // halves per (kt, nt)
    unsigned short* pWi  = packs;                  // KT=5
    unsigned short* pWh  = pWi  + (size_t)5  * NT * FRAG;   // KT=10
    unsigned short* pWoT = pWh  + (size_t)10 * NT * FRAG;   // KT=5  (K=136)
    unsigned short* pWoB = pWoT + (size_t)5  * NT * FRAG;   // KT=10 (K=300)
    unsigned short* pWa  = pWoB + (size_t)10 * NT * FRAG;   // KT=10
    unsigned short* pWb  = pWa  + (size_t)10 * NT * FRAG;   // KT=10

    float* qbuf   = binput;                        // reuse after bond loop
    float* att0   = binput + NAH;
    unsigned short* nei_a = nei;                   // reuse after bond loop
    float* tmpO   = scratch0;
    float* atom_h = scratch1;
    float* att_hb = scratch0;

    const dim3 blk(256);
    const dim3 gB(N_BONDS / 128, 2);
    const dim3 gA(N_ATOMS / 128, 2);
    auto pkb = [](int kt_) { return (kt_ * NT * 32 + 255) / 256; };

    // -------- 0) pack weights into WMMA B-fragment layout (L2-resident) --------
    pack_weights<<<pkb(5),  blk, 0, stream>>>(W_i,              pWi,  148, HIDDEN, 5);
    pack_weights<<<pkb(10), blk, 0, stream>>>(W_h,              pWh,  HIDDEN, HIDDEN, 10);
    pack_weights<<<pkb(5),  blk, 0, stream>>>(W_o,              pWoT, 136, HIDDEN, 5);
    pack_weights<<<pkb(10), blk, 0, stream>>>(W_o + 136 * HIDDEN, pWoB, HIDDEN, HIDDEN, 10);
    pack_weights<<<pkb(10), blk, 0, stream>>>(W_a,              pWa,  HIDDEN, HIDDEN, 10);
    pack_weights<<<pkb(10), blk, 0, stream>>>(W_b,              pWb,  HIDDEN, HIDDEN, 10);

    // -------- 1) binput = fbonds @ W_i (f32); message = relu -> bf16 --------
    gemm_wmma<false><<<gB, blk, 0, stream>>>(fbonds, pWi, nullptr, nullptr,
                                             binput, nullptr, message,
                                             N_BONDS, 148, HIDDEN, 5);

    // -------- 2) message passing: 5 steps --------
    const int gatherB_blocks = (int)(((long)N_BONDS * ROW_F4 + 255) / 256);
    for (int d = 0; d < 5; ++d) {
        gather_sum6_bf16<<<gatherB_blocks, blk, 0, stream>>>(message, bgraph, nei, N_BONDS);
        gemm_wmma<true><<<gB, blk, 0, stream>>>(nei, pWh, binput, nullptr,
                                                nullptr, nullptr, message,
                                                N_BONDS, HIDDEN, HIDDEN, 10);
    }

    // -------- 3) atom-side gather (bf16) --------
    const int gatherA_blocks = (int)(((long)N_ATOMS * ROW_F4 + 255) / 256);
    gather_sum6_bf16<<<gatherA_blocks, blk, 0, stream>>>(message, agraph, nei_a, N_ATOMS);

    // -------- 4) atom_h = relu([fatoms, nei_a] @ W_o + b_o), split K --------
    gemm_wmma<false><<<gA, blk, 0, stream>>>(fatoms, pWoT, nullptr, nullptr,
                                             tmpO, nullptr, nullptr,
                                             N_ATOMS, 136, HIDDEN, 5);
    gemm_wmma<true><<<gA, blk, 0, stream>>>(nei_a, pWoB, tmpO, b_o,
                                            nullptr, atom_h, nullptr,
                                            N_ATOMS, HIDDEN, HIDDEN, 10);

    // -------- 5) q = atom_h @ W_a --------
    gemm_wmma<false><<<gA, blk, 0, stream>>>(atom_h, pWa, nullptr, nullptr,
                                             qbuf, nullptr, nullptr,
                                             N_ATOMS, HIDDEN, HIDDEN, 10);

    // -------- 6) att0 = softmax(q h^T) h --------
    attention_kernel<<<N_MOLS, blk, 0, stream>>>(atom_h, qbuf, att0);

    // -------- 7) att_hb = relu(att0 @ W_b + b_b) --------
    gemm_wmma<false><<<gA, blk, 0, stream>>>(att0, pWb, nullptr, b_b,
                                             nullptr, att_hb, nullptr,
                                             N_ATOMS, HIDDEN, HIDDEN, 10);

    // -------- 8) pooled molecule vectors --------
    const int pool_blocks = (N_MOLS * ROW_F4 + 255) / 256;
    pool_kernel<<<pool_blocks, blk, 0, stream>>>(atom_h, att_hb, out);

    (void)in_sizes; (void)n_in; (void)out_size; (void)ws_size;
}
